// TrajectoryLSTM_29386166239817
// MI455X (gfx1250) — matrix operand order repaired and checked by reading, not compiled, not run on hardware
//
#include <hip/hip_runtime.h>
#include <hip/hip_bf16.h>

typedef __attribute__((ext_vector_type(16))) _Float16 v16h;
typedef __attribute__((ext_vector_type(8)))  float    v8f;

#define TT   4
#define DEMB 64
#define HH   128
#define ROWS 32   // 1 target + 16 neighbors, padded to 2 WMMA M-tiles

// Branch-free transcendentals: v_exp_f32 + v_rcp_f32 are single TRANS ops that
// co-execute with the XDL/WMMA pipe; saturation falls out of inf/rcp semantics.
__device__ __forceinline__ float fast_rcp(float x) {
  return __builtin_amdgcn_rcpf(x);
}
__device__ __forceinline__ float sigmoidf_(float x) {
  return fast_rcp(1.0f + __expf(-x));   // x->-inf: exp->inf, rcp->0
}
__device__ __forceinline__ float tanhf_(float x) {
  return 1.0f - 2.0f * fast_rcp(1.0f + __expf(2.0f * x));  // saturates to +/-1
}

// B fragment (K x 16) for D = A*B: B[k][n] = W[n0+n][k], W row-major (rows x Kd).
// CDNA5 16-bit B layout: lane%16 = N column, lane/16 = hl selects K half:
//   vgpr pair j<4 : K = hl*8 + 2j ; j>=4 : K = 16 + hl*8 + 2(j-4)
__device__ __forceinline__ v16h load_bfrag(const float* __restrict__ W, int Kd,
                                           int n0, int kbase, int lane) {
  v16h b;
  const int n = n0 + (lane & 15);
  const int h = lane >> 4;
  #pragma unroll
  for (int j = 0; j < 8; ++j) {
    const int kk = kbase + ((j < 4) ? (h * 8 + 2 * j) : (16 + h * 8 + 2 * (j - 4)));
    b[2 * j]     = (_Float16)W[(size_t)n * Kd + kk];
    b[2 * j + 1] = (_Float16)W[(size_t)n * Kd + kk + 1];
  }
  return b;
}

// A fragment (16 x 32) from an f16 row vector in LDS.
// CDNA5 16-bit A layout: lane%16 = M row, lane/16 = hl selects K half:
//   vgpr pair j<4 : K = hl*8 + 2j ; j>=4 : K = 16 + hl*8 + 2(j-4)
// Each half is 16B contiguous -> lowers to ds_load_b128.
__device__ __forceinline__ v16h load_afrag(const _Float16* __restrict__ rowp,
                                           int kbase, int hl) {
  v16h a;
  #pragma unroll
  for (int j = 0; j < 8; ++j) {
    const int kk = kbase + ((j < 4) ? (hl * 8 + 2 * j) : (16 + hl * 8 + 2 * (j - 4)));
    a[2 * j]     = rowp[kk];
    a[2 * j + 1] = rowp[kk + 1];
  }
  return a;
}

__global__ __launch_bounds__(256)
void traj_lstm_fused(const float* __restrict__ xg, const float* __restrict__ ng,
                     const int* __restrict__ maskg,
                     const float* __restrict__ W_embed, const float* __restrict__ b_embed,
                     const float* __restrict__ W_ih, const float* __restrict__ W_hh,
                     const float* __restrict__ b_ih, const float* __restrict__ b_hh,
                     const float* __restrict__ W1, const float* __restrict__ b1,
                     const float* __restrict__ W2, const float* __restrict__ b2,
                     const float* __restrict__ Wp, const float* __restrict__ bp,
                     float* __restrict__ out, int Bsz) {
  __shared__ float    xs[ROWS][TT][4];
  __shared__ int      smask[16];
  __shared__ _Float16 ex[ROWS][TT][DEMB];   // embedded inputs, f16
  __shared__ _Float16 hbuf[ROWS][HH];       // recurrent h, f16
  __shared__ float    hfin[2 * HH];
  __shared__ float    hid1[HH];

  const int tid  = threadIdx.x;
  const int lane = tid & 31;
  const int wave = tid >> 5;
  const int ego  = blockIdx.x;

  // ---- stage raw inputs for this ego (rows >16 are zero pad) ----
  for (int idx = tid; idx < ROWS * TT * 4; idx += 256) {
    const int i = idx & 3, t = (idx >> 2) & 3, row = idx >> 4;
    float v = 0.0f;
    if (row == 0)        v = xg[(size_t)ego * 16 + t * 4 + i];
    else if (row <= 16)  v = ng[(size_t)ego * 256 + (size_t)(row - 1) * 16 + t * 4 + i];
    xs[row][t][i] = v;
  }
  if (tid < 16) smask[tid] = maskg[(size_t)ego * 16 + tid];
  for (int idx = tid; idx < ROWS * HH; idx += 256)
    (&hbuf[0][0])[idx] = (_Float16)0.0f;
  __syncthreads();

  // ---- embedding: relu(x @ We^T + be) -> f16 ----
  for (int idx = tid; idx < ROWS * TT * DEMB; idx += 256) {
    const int d = idx & 63, t = (idx >> 6) & 3, row = idx >> 8;
    float a = b_embed[d];
    #pragma unroll
    for (int i = 0; i < 4; ++i) a += xs[row][t][i] * W_embed[d * 4 + i];
    ex[row][t][d] = (_Float16)fmaxf(a, 0.0f);
  }
  __syncthreads();

  // ---- LSTM: wave w owns hidden slice j0..j0+15 -> gate tiles stay wave-local ----
  const int j0 = wave * 16;
  const int nl = lane & 15;
  const int hl = lane >> 4;

  float biasv[4];
  #pragma unroll
  for (int g = 0; g < 4; ++g)
    biasv[g] = b_ih[g * HH + j0 + nl] + b_hh[g * HH + j0 + nl];

  v8f cst[2];
  cst[0] = (v8f){};
  cst[1] = (v8f){};

  for (int t = 0; t < TT; ++t) {
    v8f hnew[2];
    #pragma unroll
    for (int mt = 0; mt < 2; ++mt) {
      const int arow = mt * 16 + nl;
      const _Float16* exrow = &ex[arow][t][0];
      const _Float16* hrow  = &hbuf[arow][0];
      const v16h ax0 = load_afrag(exrow,  0, hl);
      const v16h ax1 = load_afrag(exrow, 32, hl);
      const v16h ah0 = load_afrag(hrow,   0, hl);
      const v16h ah1 = load_afrag(hrow,  32, hl);
      const v16h ah2 = load_afrag(hrow,  64, hl);
      const v16h ah3 = load_afrag(hrow,  96, hl);

      v8f accs[4];
      #pragma unroll
      for (int g = 0; g < 4; ++g) {
        const int n0 = g * HH + j0;
        v8f acc = {};
        v16h b;
        b = load_bfrag(W_ih, DEMB, n0,  0, lane);
        acc = __builtin_amdgcn_wmma_f32_16x16x32_f16(false, ax0, false, b, (short)0, acc, false, false);
        b = load_bfrag(W_ih, DEMB, n0, 32, lane);
        acc = __builtin_amdgcn_wmma_f32_16x16x32_f16(false, ax1, false, b, (short)0, acc, false, false);
        b = load_bfrag(W_hh, HH,   n0,  0, lane);
        acc = __builtin_amdgcn_wmma_f32_16x16x32_f16(false, ah0, false, b, (short)0, acc, false, false);
        b = load_bfrag(W_hh, HH,   n0, 32, lane);
        acc = __builtin_amdgcn_wmma_f32_16x16x32_f16(false, ah1, false, b, (short)0, acc, false, false);
        b = load_bfrag(W_hh, HH,   n0, 64, lane);
        acc = __builtin_amdgcn_wmma_f32_16x16x32_f16(false, ah2, false, b, (short)0, acc, false, false);
        b = load_bfrag(W_hh, HH,   n0, 96, lane);
        acc = __builtin_amdgcn_wmma_f32_16x16x32_f16(false, ah3, false, b, (short)0, acc, false, false);
        accs[g] = acc;
      }

      // LSTM cell, PyTorch gate order i,f,g,o — all wave-local, branch-free.
      v8f hn;
      #pragma unroll
      for (int r = 0; r < 8; ++r) {
        const float iv = sigmoidf_(accs[0][r] + biasv[0]);
        const float fv = sigmoidf_(accs[1][r] + biasv[1]);
        const float gv = tanhf_(accs[2][r] + biasv[2]);
        const float ov = sigmoidf_(accs[3][r] + biasv[3]);
        const float cv = fv * cst[mt][r] + iv * gv;
        cst[mt][r] = cv;
        hn[r] = ov * tanhf_(cv);
      }
      hnew[mt] = hn;
    }
    __syncthreads();  // all reads of h_{t-1} complete (WAR)
    #pragma unroll
    for (int mt = 0; mt < 2; ++mt) {
      #pragma unroll
      for (int r = 0; r < 8; ++r)
        hbuf[mt * 16 + r + 8 * hl][j0 + nl] = (_Float16)hnew[mt][r];
    }
    __syncthreads();  // h_t visible to all waves
  }

  // ---- masked mean pool + concat ----
  if (tid < HH) {
    const int j = tid;
    hfin[j] = (float)hbuf[0][j];
    float s = 0.0f, cnt = 0.0f;
    #pragma unroll
    for (int nb = 0; nb < 16; ++nb) {
      if (smask[nb]) s += (float)hbuf[1 + nb][j];
      cnt += smask[nb] ? 1.0f : 0.0f;
    }
    hfin[HH + j] = s / fmaxf(cnt, 1.0f);
  }
  __syncthreads();

  // ---- MLP head: relu(hf @ W1^T + b1) ----
  if (tid < HH) {
    float a = b1[tid];
    const float* wr = W1 + (size_t)tid * 256;
    for (int k2 = 0; k2 < 256; ++k2) a += wr[k2] * hfin[k2];
    hid1[tid] = fmaxf(a, 0.0f);
  }
  __syncthreads();

  // ---- traj = hid1 @ W2^T + b2  (B, 36) ----
  if (tid < 36) {
    float a = b2[tid];
    const float* wr = W2 + (size_t)tid * 128;
    for (int k2 = 0; k2 < 128; ++k2) a += wr[k2] * hid1[k2];
    out[(size_t)ego * 36 + tid] = a;
  }

  // ---- probs = softmax(hf @ Wp^T + bp)  (B, 3) ----
  if (tid == 64) {
    float lg[3];
    #pragma unroll
    for (int m = 0; m < 3; ++m) {
      float a = bp[m];
      const float* wr = Wp + (size_t)m * 256;
      for (int k2 = 0; k2 < 256; ++k2) a += wr[k2] * hfin[k2];
      lg[m] = a;
    }
    const float mx = fmaxf(lg[0], fmaxf(lg[1], lg[2]));
    const float e0 = __expf(lg[0] - mx), e1 = __expf(lg[1] - mx), e2 = __expf(lg[2] - mx);
    const float den = e0 + e1 + e2;
    float* pr = out + (size_t)Bsz * 36 + (size_t)ego * 3;
    pr[0] = e0 / den; pr[1] = e1 / den; pr[2] = e2 / den;
  }
}

extern "C" void kernel_launch(void* const* d_in, const int* in_sizes, int n_in,
                              void* d_out, int out_size, void* d_ws, size_t ws_size,
                              hipStream_t stream) {
  const float* xg      = (const float*)d_in[0];
  const float* ng      = (const float*)d_in[1];
  const int*   maskg   = (const int*)d_in[2];
  const float* W_embed = (const float*)d_in[3];
  const float* b_embed = (const float*)d_in[4];
  const float* W_ih    = (const float*)d_in[5];
  const float* W_hh    = (const float*)d_in[6];
  const float* b_ih    = (const float*)d_in[7];
  const float* b_hh    = (const float*)d_in[8];
  const float* W1      = (const float*)d_in[9];
  const float* b1      = (const float*)d_in[10];
  const float* W2      = (const float*)d_in[11];
  const float* b2      = (const float*)d_in[12];
  const float* Wp      = (const float*)d_in[13];
  const float* bp      = (const float*)d_in[14];
  float* out = (float*)d_out;

  const int Bsz = in_sizes[0] / (TT * 4);  // x is (B, T, 4)
  hipLaunchKernelGGL(traj_lstm_fused, dim3(Bsz), dim3(256), 0, stream,
                     xg, ng, maskg, W_embed, b_embed, W_ih, W_hh, b_ih, b_hh,
                     W1, b1, W2, b2, Wp, bp, out, Bsz);
}
